// MinGRU_54898271977819
// MI455X (gfx1250) — matrix operand order repaired and checked by reading, compiled, not run on hardware
//
#include <hip/hip_runtime.h>
#include <cstdint>
#include <cstddef>

// Problem dims (fixed by reference)
#define Bdim 16
#define Tdim 2048
#define Ddim 1024
#define Hdim 1024
#define Mdim (Bdim * Tdim)   // 32768 rows
#define Kdim Ddim            // 1024
#define BHdim (Bdim * Hdim)  // 16384 independent scan sequences
#define NCH 16               // time chunks
#define CHLEN (Tdim / NCH)   // 128 steps per chunk
#define KSTAGES (Kdim / 32)  // 32 K-stages of 32 columns

// POD vector types (avoid HIP's non-trivial vector classes inside unions)
typedef __attribute__((ext_vector_type(16))) __bf16       v16bf;
typedef __attribute__((ext_vector_type(8)))  float        v8f;
typedef __attribute__((ext_vector_type(4)))  unsigned int u32x4;
typedef __attribute__((ext_vector_type(2)))  unsigned int u32x2;
typedef __attribute__((ext_vector_type(4)))  float        f32x4;

union FragU { v16bf v; u32x4 q[2]; };

__device__ __forceinline__ unsigned int bf16_rn(float f) {
  unsigned int u = __float_as_uint(f);
  return (u + 0x7fffu + ((u >> 16) & 1u)) >> 16;  // round-to-nearest-even
}

// Low 32 bits of a generic pointer to __shared__ == LDS byte offset (the LDS
// aperture base has zero low bits; HW uses addr[31:0] as the DS address).
__device__ __forceinline__ unsigned lds_lo32(const void* p) {
  return (unsigned)(unsigned long long)(uintptr_t)p;
}

// Async global -> LDS 128-bit copy (CDNA5, tracked by ASYNCcnt).
// GVS addressing: mem = SADDR(64) + VADDR(32); LDS dest = VDST lane value.
__device__ __forceinline__ void async_b128(unsigned ldsAddr, unsigned gOff,
                                           unsigned long long base) {
  asm volatile("global_load_async_to_lds_b128 %0, %1, %2"
               :: "v"(ldsAddr), "v"(gOff), "s"(base) : "memory");
}

// ---------------------------------------------------------------------------
// fp32 -> bf16 pack (4 elems/thread, 128-bit load, 64-bit store)
// ---------------------------------------------------------------------------
__global__ void cvt_f32_bf16(const float* __restrict__ in,
                             unsigned int* __restrict__ out, int n4) {
  int i = blockIdx.x * blockDim.x + threadIdx.x;
  if (i >= n4) return;
  f32x4 f = ((const f32x4*)in)[i];
  u32x2 o;
  o.x = bf16_rn(f.x) | (bf16_rn(f.y) << 16);
  o.y = bf16_rn(f.z) | (bf16_rn(f.w) << 16);
  ((u32x2*)out)[i] = o;
}

// ---------------------------------------------------------------------------
// Fused gate GEMM: each 128(M) x 128(H) tile computes BOTH gate halves
// (W rows [h, h+128) and [H+h, H+h+128)) sharing the x tile in LDS.
// K-step 32, double-buffered LDS filled by async global->LDS (ASYNCcnt),
// bf16 WMMA accumulate in f32, fused sigmoid epilogue:
//   a = 1 - sigmoid(zp) -> ws, bb = sigmoid(zp)*sigmoid(tp) -> d_out.
// Block = 256 threads = 8 waves; wave tile = 32(M) x 64(N) per gate half.
// Steady state (fully unrolled, branch-free): issue stage s+1 async fills,
// s_wait_asynccnt 6 (stage s retired, in-order), barrier, 16 WMMA, barrier.
// ---------------------------------------------------------------------------
__global__ __launch_bounds__(256)
void mingru_gemm_gate(const u32x4* __restrict__ xbf,   // [M][K/8] (bf16 pairs)
                      const u32x4* __restrict__ wbf,   // [2H][K/8]
                      const float* __restrict__ bias,  // [2H]
                      float* __restrict__ aOut,        // [M][H] -> ws
                      float* __restrict__ bbOut)       // [M][H] -> d_out
{
  // 128 rows x 32 bf16 per tile (8 KB), double buffered: 3 * 16 KB = 48 KB
  __shared__ u32x4 sA [2][512];
  __shared__ u32x4 sBz[2][512];
  __shared__ u32x4 sBt[2][512];

  const int tid     = threadIdx.x;
  const int lane    = tid & 31;
  const int wave    = tid >> 5;
  const int waveM   = wave >> 1;   // 0..3  -> 32-row slab
  const int waveN   = wave & 1;    // 0..1  -> 64-col slab
  const int l15     = lane & 15;
  const int halfSel = lane >> 4;   // 0|1 selects K-half (contiguous k = 16*halfSel + s)

  const int mBase = blockIdx.x * 128;
  const int hBase = blockIdx.y * 128;

  const unsigned long long baseX = (unsigned long long)(uintptr_t)xbf;
  const unsigned long long baseW = (unsigned long long)(uintptr_t)wbf;
  const unsigned sAo  = lds_lo32(&sA[0][0]);
  const unsigned sBzo = lds_lo32(&sBz[0][0]);
  const unsigned sBto = lds_lo32(&sBt[0][0]);

  // Per-thread tile-fill invariants: thread covers 2 u32x4 slots per tile.
  const int r0 = tid >> 2, c0 = tid & 3;              // slot 0: idx = tid
  const int r1 = (tid + 256) >> 2, c1 = tid & 3;      // slot 1: idx = tid+256
  const unsigned gx0 = (unsigned)((mBase + r0) * 2048 + c0 * 16);
  const unsigned gx1 = (unsigned)((mBase + r1) * 2048 + c1 * 16);
  const unsigned gw0 = (unsigned)((hBase + r0) * 2048 + c0 * 16);
  const unsigned gw1 = (unsigned)((hBase + r1) * 2048 + c1 * 16);
  const unsigned ld0 = (unsigned)tid * 16u;
  const unsigned ld1 = (unsigned)(tid + 256) * 16u;

  v8f accZ[2][4], accT[2][4];
  const v8f vzero = {0.f, 0.f, 0.f, 0.f, 0.f, 0.f, 0.f, 0.f};
#pragma unroll
  for (int m = 0; m < 2; ++m)
#pragma unroll
    for (int n = 0; n < 4; ++n) { accZ[m][n] = vzero; accT[m][n] = vzero; }

  // Issue one K-stage (32 columns) of async tile fills: 6 ops per thread.
  auto issue = [&](int bufSel, int stage) {
    const unsigned kb = (unsigned)stage * 64u;        // 32 bf16 = 64 bytes
    const unsigned bo = (unsigned)bufSel * 8192u;
    async_b128(sAo  + bo + ld0, gx0 + kb, baseX);
    async_b128(sBzo + bo + ld0, gw0 + kb, baseW);
    async_b128(sBto + bo + ld0, gw0 + kb + 2097152u, baseW);
    async_b128(sAo  + bo + ld1, gx1 + kb, baseX);
    async_b128(sBzo + bo + ld1, gw1 + kb, baseW);
    async_b128(sBto + bo + ld1, gw1 + kb + 2097152u, baseW);
  };

  // Consume one K-stage from LDS buffer `bufSel`: 20 ds_load_b128 + 16 WMMA.
  auto compute = [&](int bufSel) {
    const u32x4* pA  = &sA [bufSel][0];
    const u32x4* pBz = &sBz[bufSel][0];
    const u32x4* pBt = &sBt[bufSel][0];
    // A frags: lane l15 = row M; halfSel picks K 0..15 vs 16..31
    FragU aF[2];
#pragma unroll
    for (int m = 0; m < 2; ++m) {
      int row = waveM * 32 + m * 16 + l15;
      int q0  = row * 4 + halfSel * 2;
      aF[m].q[0] = pA[q0];
      aF[m].q[1] = pA[q0 + 1];
    }
#pragma unroll
    for (int n = 0; n < 4; ++n) {
      // B frags: lane l15 = column N; halfSel picks K 0..15 vs 16..31
      int wrow = waveN * 64 + n * 16 + l15;
      int q0   = wrow * 4 + halfSel * 2;
      FragU bZ, bT;
      bZ.q[0] = pBz[q0]; bZ.q[1] = pBz[q0 + 1];
      bT.q[0] = pBt[q0]; bT.q[1] = pBt[q0 + 1];
#pragma unroll
      for (int m = 0; m < 2; ++m) {
        accZ[m][n] = __builtin_amdgcn_wmma_f32_16x16x32_bf16(
            false, aF[m].v, false, bZ.v, (short)0, accZ[m][n], false, false);
        accT[m][n] = __builtin_amdgcn_wmma_f32_16x16x32_bf16(
            false, aF[m].v, false, bT.v, (short)0, accT[m][n], false, false);
      }
    }
  };

  issue(0, 0);  // prologue: stage 0 in flight

  // Steady state: branch-free, buffer index is a compile-time constant.
#pragma unroll
  for (int s = 0; s < KSTAGES - 1; ++s) {
    issue((s + 1) & 1, s + 1);                        // next stage in flight
    asm volatile("s_wait_asynccnt 0x6" ::: "memory"); // stage s retired
    __syncthreads();                                  // fills visible to all
    compute(s & 1);
    __syncthreads();                                  // safe to refill buf
  }
  // Tail: last stage, nothing left in flight.
  asm volatile("s_wait_asynccnt 0x0" ::: "memory");
  __syncthreads();
  compute((KSTAGES - 1) & 1);

  // Fused epilogue: bias + sigmoid gating, write a (ws) and bb (d_out)
#pragma unroll
  for (int n = 0; n < 4; ++n) {
    int col  = hBase + waveN * 64 + n * 16 + l15;
    float bz = bias[col];
    float bt = bias[Hdim + col];
#pragma unroll
    for (int m = 0; m < 2; ++m) {
      int rb = mBase + waveM * 32 + m * 16 + halfSel * 8;  // M = e + 8*halfSel
#pragma unroll
      for (int e = 0; e < 8; ++e) {
        float zp = accZ[m][n][e] + bz;
        float tp = accT[m][n][e] + bt;
        float z  = 1.0f / (1.0f + __expf(-zp));
        float tl = 1.0f / (1.0f + __expf(-tp));
        size_t o = (size_t)(rb + e) * Hdim + col;
        aOut[o]  = 1.0f - z;
        bbOut[o] = z * tl;
      }
    }
  }
}

// ---------------------------------------------------------------------------
// Scan phase 1: per-(b,h,chunk) carry composition (A,B): h_out = A*h_in + B
// ---------------------------------------------------------------------------
__global__ void mingru_scan_chunk(const float* __restrict__ aBuf,
                                  const float* __restrict__ bbBuf,
                                  float* __restrict__ cA,
                                  float* __restrict__ cB) {
  int tid = blockIdx.x * 256 + threadIdx.x;
  int bh  = tid & (BHdim - 1);
  int c   = tid >> 14;
  int b   = bh >> 10;
  int h   = bh & (Hdim - 1);
  size_t base = (size_t)b * Tdim * Hdim + (size_t)(c * CHLEN) * Hdim + h;
  float A = 1.0f, Bv = 0.0f;
  for (int t = 0; t < CHLEN; ++t) {
    float a  = aBuf[base];
    float bb = bbBuf[base];
    A  = a * A;
    Bv = __fmaf_rn(a, Bv, bb);
    base += Hdim;
  }
  cA[(size_t)c * BHdim + bh] = A;
  cB[(size_t)c * BHdim + bh] = Bv;
}

// ---------------------------------------------------------------------------
// Scan phase 2: exclusive prefix over the 16 chunk carries, seeded with h0
// ---------------------------------------------------------------------------
__global__ void mingru_scan_carry(const float* __restrict__ cA,
                                  const float* __restrict__ cB,
                                  const float* __restrict__ h0,
                                  float* __restrict__ hIn) {
  int bh = blockIdx.x * 256 + threadIdx.x;
  float s = h0[bh];
  for (int c = 0; c < NCH; ++c) {
    hIn[(size_t)c * BHdim + bh] = s;
    s = __fmaf_rn(cA[(size_t)c * BHdim + bh], s, cB[(size_t)c * BHdim + bh]);
  }
}

// ---------------------------------------------------------------------------
// Scan phase 3: apply recurrence inside each chunk, in place on d_out (bb->h)
// ---------------------------------------------------------------------------
__global__ void mingru_scan_apply(const float* __restrict__ aBuf,
                                  const float* __restrict__ hIn,
                                  float* __restrict__ out) {
  int tid = blockIdx.x * 256 + threadIdx.x;
  int bh  = tid & (BHdim - 1);
  int c   = tid >> 14;
  int b   = bh >> 10;
  int h   = bh & (Hdim - 1);
  size_t base = (size_t)b * Tdim * Hdim + (size_t)(c * CHLEN) * Hdim + h;
  float hv = hIn[(size_t)c * BHdim + bh];
  for (int t = 0; t < CHLEN; ++t) {
    float a  = aBuf[base];
    float bb = out[base];
    hv = __fmaf_rn(a, hv, bb);
    out[base] = hv;
    base += Hdim;
  }
}

// ---------------------------------------------------------------------------
// Launcher. Inputs: x [B,T,D] f32, h0 [B,H] f32, W [2H,D] f32, b [2H] f32.
// Output: h [B,T,H] f32. Workspace layout (bytes):
//   [0, 128M)         aBuf   (a = 1-z, fp32, B*T*H)
//   [128M, 192M)      xbf    (x as bf16)
//   [192M, 196M)      wbf    (W as bf16)
//   [196M .. +3M)     chunk carries cA, cB, hIn
// ---------------------------------------------------------------------------
extern "C" void kernel_launch(void* const* d_in, const int* in_sizes, int n_in,
                              void* d_out, int out_size, void* d_ws, size_t ws_size,
                              hipStream_t stream) {
  (void)in_sizes; (void)n_in; (void)out_size; (void)ws_size;
  const float* x  = (const float*)d_in[0];
  const float* h0 = (const float*)d_in[1];
  const float* W  = (const float*)d_in[2];
  const float* b  = (const float*)d_in[3];
  float* out = (float*)d_out;

  char* ws = (char*)d_ws;
  float*        aBuf = (float*)ws;
  unsigned int* xbf  = (unsigned int*)(ws + (size_t)134217728);
  unsigned int* wbf  = (unsigned int*)(ws + (size_t)201326592);
  float*        cA   = (float*)(ws + (size_t)205520896);
  float*        cB   = (float*)(ws + (size_t)206569472);
  float*        hIn  = (float*)(ws + (size_t)207618048);

  // 1) fp32 -> bf16 packing (x and W)
  cvt_f32_bf16<<<(Mdim * Kdim / 4) / 256, 256, 0, stream>>>(x, xbf, Mdim * Kdim / 4);
  cvt_f32_bf16<<<(2 * Hdim * Kdim / 4) / 256, 256, 0, stream>>>(W, wbf, 2 * Hdim * Kdim / 4);

  // 2) fused bf16 WMMA GEMM + sigmoid gating (a -> ws, bb -> d_out)
  dim3 gg(Mdim / 128, Hdim / 128);  // 256 x 8 blocks
  mingru_gemm_gate<<<gg, 256, 0, stream>>>((const u32x4*)xbf, (const u32x4*)wbf,
                                           b, aBuf, out);

  // 3) two-level associative scan over T
  mingru_scan_chunk<<<(BHdim * NCH) / 256, 256, 0, stream>>>(aBuf, out, cA, cB);
  mingru_scan_carry<<<BHdim / 256, 256, 0, stream>>>(cA, cB, h0, hIn);
  mingru_scan_apply<<<(BHdim * NCH) / 256, 256, 0, stream>>>(aBuf, hIn, out);
}
